// GRUMemoryNetworkCell_54975581388889
// MI455X (gfx1250) — matrix operand order repaired and checked by reading, not compiled, not run on hardware
//
#include <hip/hip_runtime.h>

// ---------------------------------------------------------------------------
// Types for CDNA5 WMMA (wave32, v_wmma_f32_16x16x32_bf16)
// ---------------------------------------------------------------------------
typedef __bf16 bf16_t;
typedef __attribute__((ext_vector_type(16))) __bf16 v16bf;
typedef __attribute__((ext_vector_type(8)))  __bf16 v8bf;
typedef __attribute__((ext_vector_type(4)))  __bf16 v4bf;
typedef __attribute__((ext_vector_type(8)))  float  v8f;

static constexpr int kB = 512;   // batch
static constexpr int kS = 512;   // tape length
static constexpr int kI = 1024;  // input dim
static constexpr int kH = 1024;  // hidden dim
static constexpr int kA = 512;   // attention dim

// ---------------------------------------------------------------------------
// Fragment loaders (layouts per CDNA5 ISA 7.12.2)
// A 16x32 bf16: lanes 0-15 -> M=lane, kh=0; lanes 16-31 -> kh=8.
// Per lane: K = {kh..kh+7, 16+kh..16+kh+7}.
// ---------------------------------------------------------------------------
__device__ __forceinline__ v16bf load_frag_a16(const bf16_t* tile, int ld, int lane) {
    const int m  = lane & 15;
    const int kh = (lane & 16) ? 8 : 0;
    const bf16_t* p = tile + m * ld + kh;
    v8bf lo = *(const v8bf*)(p);
    v8bf hi = *(const v8bf*)(p + 16);
    v16bf r;
#pragma unroll
    for (int i = 0; i < 8; ++i) { r[i] = lo[i]; r[i + 8] = hi[i]; }
    return r;
}

// B 32x16 bf16 (B = W^T, W row-major [N,K]): lane n = lane&15,
// lanes 0-15 kh=0, lanes 16-31 kh=8; per lane K = {kh..kh+7, 16+kh..16+kh+7}.
__device__ __forceinline__ v16bf load_frag_b16(const bf16_t* W, int ld, int n0, int k0, int lane) {
    const int n  = lane & 15;
    const int kh = (lane & 16) ? 8 : 0;
    const bf16_t* p = W + (size_t)(n0 + n) * ld + k0 + kh;
    v8bf lo = *(const v8bf*)(p);
    v8bf hi = *(const v8bf*)(p + 16);
    v16bf r;
#pragma unroll
    for (int i = 0; i < 8; ++i) { r[i] = lo[i]; r[i + 8] = hi[i]; }
    return r;
}

__device__ __forceinline__ v8f wmma_bf16(v16bf a, v16bf b, v8f c) {
    return __builtin_amdgcn_wmma_f32_16x16x32_bf16(
        /*neg_a=*/false, a, /*neg_b=*/false, b,
        /*c_mod=*/(short)0, c, /*reuse_a=*/false, /*reuse_b=*/false);
}

// ---------------------------------------------------------------------------
// f32 -> bf16 conversion
// ---------------------------------------------------------------------------
__global__ __launch_bounds__(256) void f32_to_bf16_kernel(const float* __restrict__ src,
                                                          bf16_t* __restrict__ dst, int n) {
    int i = blockIdx.x * 256 + threadIdx.x;
    if (i < n) dst[i] = (bf16_t)src[i];
}

// bias3[a] = bWx[a] + bWth[a] + bWh[a]
__global__ __launch_bounds__(256) void biasvec_kernel(const float* __restrict__ bWx,
                                                      const float* __restrict__ bWth,
                                                      const float* __restrict__ bWh,
                                                      float* __restrict__ out, int n) {
    int i = blockIdx.x * 256 + threadIdx.x;
    if (i < n) out[i] = bWx[i] + bWth[i] + bWh[i];
}

// ---------------------------------------------------------------------------
// Generic WMMA GEMM: C[M,N] = A[M,K] * W[N,K]^T  (+ biasN) (+ C if accumulate)
// One wave per 16x64 C strip (4 N-tiles share each A fragment).
// ---------------------------------------------------------------------------
__global__ __launch_bounds__(32) void gemm_bf16_kernel(const bf16_t* __restrict__ A,
                                                       const bf16_t* __restrict__ W,
                                                       const float* __restrict__ biasN,
                                                       float* __restrict__ C,
                                                       int M, int N, int K, int accumulate) {
    const int strips_n = N >> 6;                 // N/64
    const int tn = blockIdx.x % strips_n;
    const int tm = blockIdx.x / strips_n;
    const int m0 = tm << 4, n0 = tn << 6;
    const int lane = threadIdx.x & 31;

    v8f acc[4];
#pragma unroll
    for (int j = 0; j < 4; ++j) acc[j] = (v8f){};

    for (int k0 = 0; k0 < K; k0 += 32) {
        v16bf af = load_frag_a16(A + (size_t)m0 * K + k0, K, lane);
#pragma unroll
        for (int j = 0; j < 4; ++j) {
            v16bf bf = load_frag_b16(W, K, n0 + j * 16, k0, lane);
            acc[j] = wmma_bf16(af, bf, acc[j]);
        }
    }

    const int n  = lane & 15;
    const int mo = (lane & 16) ? 8 : 0;
#pragma unroll
    for (int j = 0; j < 4; ++j) {
        const int nc = n0 + j * 16 + n;
        const float bn = biasN ? biasN[nc] : 0.0f;
#pragma unroll
        for (int r = 0; r < 8; ++r) {
            size_t idx = (size_t)(m0 + mo + r) * N + nc;
            float v = acc[j][r] + bn;
            if (accumulate) v += C[idx];
            C[idx] = v;
        }
    }
}

// ---------------------------------------------------------------------------
// Fused attention-score kernel:
//   scores[b,s] = bv + sum_a v[a] * tanh( sum_h tape[b,s,h]*Wh[a,h] + biasBA[b,a] )
// 4 waves/block, each wave owns 16 rows of (b,s). Tape tile converted to bf16
// into LDS once (tape read exactly once from HBM). 4 N-tiles per K pass so
// each LDS A fragment feeds 4 WMMAs. Wh streamed from L2 (+L0 prefetch).
// ---------------------------------------------------------------------------
__global__ __launch_bounds__(128) void attn_scores_kernel(const float* __restrict__ tape,    // [B*S, H]
                                                          const bf16_t* __restrict__ WhB,    // [A, H] bf16
                                                          const float* __restrict__ biasBA,  // [B, A]
                                                          const float* __restrict__ vvec,    // [A]
                                                          const float* __restrict__ bv,      // [1]
                                                          float* __restrict__ scores) {      // [B*S]
    extern __shared__ bf16_t lds[];
    const int lane = threadIdx.x & 31;
    const int wave = threadIdx.x >> 5;
    const size_t row0 = (size_t)blockIdx.x * 64 + (size_t)wave * 16;
    bf16_t* myA = lds + (size_t)wave * 16 * kH;   // 16 rows x 1024 bf16 = 32 KB

    // Stage + convert: 16 rows x 1024 f32 -> bf16 in LDS (float4 granularity)
    for (int idx = lane; idx < 16 * (kH / 4); idx += 32) {
        const int r  = idx >> 8;            // idx / 256
        const int c4 = (idx & 255) << 2;    // 4-float column
        const float4 f = *(const float4*)(tape + (row0 + r) * kH + c4);
        v4bf h;
        h[0] = (bf16_t)f.x; h[1] = (bf16_t)f.y; h[2] = (bf16_t)f.z; h[3] = (bf16_t)f.w;
        *(v4bf*)(myA + r * kH + c4) = h;
    }
    __syncthreads();

    const int b = (int)(row0 >> 9);         // row0 / S, constant per wave tile
    const float bvv = bv[0];
    float sacc[8];
#pragma unroll
    for (int r = 0; r < 8; ++r) sacc[r] = 0.0f;

    for (int n0 = 0; n0 < kA; n0 += 64) {
        // Warm L0 for the next Wh N-block (gfx1250 global_prefetch_b8 path).
        if (n0 + 64 < kA)
            __builtin_prefetch((const void*)(WhB + (size_t)(n0 + 64 + (lane & 15)) * kH), 0, 1);

        v8f acc[4];
#pragma unroll
        for (int j = 0; j < 4; ++j) acc[j] = (v8f){};

#pragma unroll 2
        for (int k0 = 0; k0 < kH; k0 += 32) {
            v16bf af = load_frag_a16(myA + k0, kH, lane);
#pragma unroll
            for (int j = 0; j < 4; ++j) {
                v16bf bf = load_frag_b16(WhB, kH, n0 + j * 16, k0, lane);
                acc[j] = wmma_bf16(af, bf, acc[j]);
            }
        }

        const int n = lane & 15;
#pragma unroll
        for (int j = 0; j < 4; ++j) {
            const int nc = n0 + j * 16 + n;
            const float vval = vvec[nc];
            const float bia  = biasBA[(size_t)b * kA + nc];
#pragma unroll
            for (int r = 0; r < 8; ++r) {
                sacc[r] += tanhf(acc[j][r] + bia) * vval;
            }
        }
    }

    // Reduce over the 16 lanes that share a row M (butterfly within half-wave).
#pragma unroll
    for (int r = 0; r < 8; ++r) {
        float x = sacc[r];
        x += __shfl_xor(x, 1, 16);
        x += __shfl_xor(x, 2, 16);
        x += __shfl_xor(x, 4, 16);
        x += __shfl_xor(x, 8, 16);
        sacc[r] = x;
    }
    if ((lane & 15) == 0) {
        const int mo = (lane & 16) ? 8 : 0;
#pragma unroll
        for (int r = 0; r < 8; ++r) {
            scores[row0 + mo + r] = sacc[r] + bvv;
        }
    }
}

// ---------------------------------------------------------------------------
// Softmax over S per batch row (in place). One block per b.
// ---------------------------------------------------------------------------
__global__ __launch_bounds__(256) void softmax_kernel(float* __restrict__ scores) {
    __shared__ float red[256];
    const int b = blockIdx.x;
    const int t = threadIdx.x;
    float* row = scores + (size_t)b * kS;
    float a0 = row[t], a1 = row[t + 256];

    float m = fmaxf(a0, a1);
    red[t] = m; __syncthreads();
    for (int s = 128; s > 0; s >>= 1) { if (t < s) red[t] = fmaxf(red[t], red[t + s]); __syncthreads(); }
    m = red[0]; __syncthreads();

    float e0 = __expf(a0 - m), e1 = __expf(a1 - m);
    red[t] = e0 + e1; __syncthreads();
    for (int s = 128; s > 0; s >>= 1) { if (t < s) red[t] += red[t + s]; __syncthreads(); }
    const float inv = 1.0f / red[0];

    row[t] = e0 * inv;
    row[t + 256] = e1 * inv;
}

// ---------------------------------------------------------------------------
// aggregated[b,h] = sum_s attn[b,s] * tape[b,s,h]   (HBM streaming, float4)
// One block per b; 256 threads x float4 cover H=1024.
// Writes f32 (second output) and bf16 copy (for gh GEMM).
// ---------------------------------------------------------------------------
__global__ __launch_bounds__(256) void aggregate_kernel(const float* __restrict__ attn,
                                                        const float* __restrict__ tape,
                                                        float* __restrict__ aggOut,
                                                        bf16_t* __restrict__ aggBf) {
    const int b = blockIdx.x;
    const int t = threadIdx.x;
    const float* arow = attn + (size_t)b * kS;
    const float* trow = tape + (size_t)b * kS * kH + t * 4;
    float ax = 0.f, ay = 0.f, az = 0.f, aw = 0.f;
    for (int s = 0; s < kS; ++s) {
        const float a = arow[s];
        const float4 x = *(const float4*)(trow + (size_t)s * kH);
        ax += a * x.x; ay += a * x.y; az += a * x.z; aw += a * x.w;
    }
    const size_t o = (size_t)b * kH + t * 4;
    float4 out; out.x = ax; out.y = ay; out.z = az; out.w = aw;
    *(float4*)(aggOut + o) = out;
    v4bf hb; hb[0] = (bf16_t)ax; hb[1] = (bf16_t)ay; hb[2] = (bf16_t)az; hb[3] = (bf16_t)aw;
    *(v4bf*)(aggBf + o) = hb;
}

// ---------------------------------------------------------------------------
// GRU gate combine:
//   r = sigm(gi_r + gh_r); u = sigm(gi_u + gh_u); n = tanh(gi_n + r*gh_n)
//   new_h = (1-u)*n + u*agg
// ---------------------------------------------------------------------------
__global__ __launch_bounds__(256) void gru_final_kernel(const float* __restrict__ gi,   // [B,3H]
                                                        const float* __restrict__ gh,   // [B,3H]
                                                        const float* __restrict__ agg,  // [B,H]
                                                        float* __restrict__ newh) {     // [B,H]
    const int idx = blockIdx.x * 256 + threadIdx.x;
    if (idx >= kB * kH) return;
    const int b = idx / kH;
    const int h = idx - b * kH;
    const size_t base = (size_t)b * 3 * kH + h;
    const float ir = gi[base], iu = gi[base + kH], in_ = gi[base + 2 * kH];
    const float hr = gh[base], hu = gh[base + kH], hn = gh[base + 2 * kH];
    const float r = 1.0f / (1.0f + __expf(-(ir + hr)));
    const float u = 1.0f / (1.0f + __expf(-(iu + hu)));
    const float n = tanhf(in_ + r * hn);
    newh[idx] = (1.0f - u) * n + u * agg[idx];
}

// ---------------------------------------------------------------------------
// Host launcher
// ---------------------------------------------------------------------------
extern "C" void kernel_launch(void* const* d_in, const int* in_sizes, int n_in,
                              void* d_out, int out_size, void* d_ws, size_t ws_size,
                              hipStream_t stream) {
    const float* input  = (const float*)d_in[0];   // [B,I]
    const float* hidden = (const float*)d_in[1];   // [B,H]
    const float* tape   = (const float*)d_in[2];   // [B,S,H]
    const float* Wih    = (const float*)d_in[3];   // [3H,I]
    const float* bih    = (const float*)d_in[4];   // [3H]
    const float* Whh    = (const float*)d_in[5];   // [3H,H]
    const float* bhh    = (const float*)d_in[6];   // [3H]
    const float* Wh     = (const float*)d_in[7];   // [A,H]
    const float* bWh    = (const float*)d_in[8];   // [A]
    const float* Wx     = (const float*)d_in[9];   // [A,I]
    const float* bWx    = (const float*)d_in[10];  // [A]
    const float* Wth    = (const float*)d_in[11];  // [A,H]
    const float* bWth   = (const float*)d_in[12];  // [A]
    const float* vvec   = (const float*)d_in[13];  // [1,A]
    const float* bv     = (const float*)d_in[14];  // [1]

    float* out_newh = (float*)d_out;               // [B,H]
    float* out_agg  = (float*)d_out + (size_t)kB * kH;

    // --- workspace carve-out (256B aligned) ---
    char* ws = (char*)d_ws;
    size_t off = 0;
    auto carve = [&](size_t bytes) { char* p = ws + off; off += (bytes + 255) & ~(size_t)255; return p; };
    bf16_t* WhB   = (bf16_t*)carve((size_t)kA * kH * 2);
    bf16_t* WxB   = (bf16_t*)carve((size_t)kA * kI * 2);
    bf16_t* WthB  = (bf16_t*)carve((size_t)kA * kH * 2);
    bf16_t* WihB  = (bf16_t*)carve((size_t)3 * kH * kI * 2);
    bf16_t* WhhB  = (bf16_t*)carve((size_t)3 * kH * kH * 2);
    bf16_t* inB   = (bf16_t*)carve((size_t)kB * kI * 2);
    bf16_t* hidB  = (bf16_t*)carve((size_t)kB * kH * 2);
    bf16_t* aggB  = (bf16_t*)carve((size_t)kB * kH * 2);
    float*  bias3 = (float*)carve((size_t)kA * 4);
    float*  biasBA= (float*)carve((size_t)kB * kA * 4);
    float*  scores= (float*)carve((size_t)kB * kS * 4);
    float*  gi    = (float*)carve((size_t)kB * 3 * kH * 4);
    float*  gh    = (float*)carve((size_t)kB * 3 * kH * 4);
    (void)ws_size; (void)in_sizes; (void)n_in; (void)out_size;

    auto cvt = [&](const float* s, bf16_t* d, int n) {
        f32_to_bf16_kernel<<<(n + 255) / 256, 256, 0, stream>>>(s, d, n);
    };
    cvt(Wh,  WhB,  kA * kH);
    cvt(Wx,  WxB,  kA * kI);
    cvt(Wth, WthB, kA * kH);
    cvt(Wih, WihB, 3 * kH * kI);
    cvt(Whh, WhhB, 3 * kH * kH);
    cvt(input,  inB,  kB * kI);
    cvt(hidden, hidB, kB * kH);

    biasvec_kernel<<<(kA + 255) / 256, 256, 0, stream>>>(bWx, bWth, bWh, bias3, kA);

    // biasBA = input @ Wx^T + (bWx+bWth+bWh);  biasBA += hidden @ Wth^T
    gemm_bf16_kernel<<<(kB / 16) * (kA / 64), 32, 0, stream>>>(inB,  WxB,  bias3,   biasBA, kB, kA, kI, 0);
    gemm_bf16_kernel<<<(kB / 16) * (kA / 64), 32, 0, stream>>>(hidB, WthB, nullptr, biasBA, kB, kA, kH, 1);

    // Fused attention scores: 4 waves/block, 64 rows/block, 128 KB dynamic LDS
    {
        const int blocks = (kB * kS) / 64;
        const size_t shmem = (size_t)4 * 16 * kH * sizeof(bf16_t); // 131072
        attn_scores_kernel<<<blocks, 128, shmem, stream>>>(tape, WhB, biasBA, vvec, bv, scores);
    }

    softmax_kernel<<<kB, 256, 0, stream>>>(scores);

    aggregate_kernel<<<kB, 256, 0, stream>>>(scores, tape, out_agg, aggB);

    // gi = input @ Wih^T + bih ; gh = aggregated @ Whh^T + bhh
    gemm_bf16_kernel<<<(kB / 16) * ((3 * kH) / 64), 32, 0, stream>>>(inB,  WihB, bih, gi, kB, 3 * kH, kI, 0);
    gemm_bf16_kernel<<<(kB / 16) * ((3 * kH) / 64), 32, 0, stream>>>(aggB, WhhB, bhh, gh, kB, 3 * kH, kH, 0);

    gru_final_kernel<<<(kB * kH + 255) / 256, 256, 0, stream>>>(gi, gh, out_agg, out_newh);
}
